// GraphNetBlock_57062935495422
// MI455X (gfx1250) — matrix operand order repaired and checked by reading, not compiled, not run on hardware
//
#include <hip/hip_runtime.h>

#define DD 128

typedef __attribute__((ext_vector_type(16))) _Float16 v16h;
typedef __attribute__((ext_vector_type(8)))  _Float16 v8h;
typedef __attribute__((ext_vector_type(8)))  float    v8f;
typedef __attribute__((ext_vector_type(4)))  float    v4f;

// ---------------- utility kernels ----------------

__global__ __launch_bounds__(256) void zero_f4(float4* p, int n4) {
  int i = blockIdx.x * 256 + threadIdx.x;
  if (i < n4) p[i] = make_float4(0.f, 0.f, 0.f, 0.f);
}

// Reformat W [K x 128] (f32, row-major) into WMMA B-fragment order:
// packed[((kc*8 + ntile)*32 + lane)*16 + i] = (f16) W[kc*32 + (i&8)*2 + (lane>>4)*8 + (i&7)][ntile*16 + (lane&15)]
__global__ __launch_bounds__(256) void pack_w(const float* __restrict__ W,
                                              _Float16* __restrict__ P, int total) {
  int idx = blockIdx.x * 256 + threadIdx.x;
  if (idx >= total) return;
  int i    = idx & 15;
  int lane = (idx >> 4) & 31;
  int nt   = (idx >> 9) & 7;
  int kc   = idx >> 12;
  int lrow = lane & 15, hs = lane >> 4;
  int k = kc * 32 + (i & 8) * 2 + hs * 8 + (i & 7);
  int c = nt * 16 + lrow;
  P[idx] = (_Float16)W[k * DD + c];
}

// Native no-return f32 atomic add at device scope (L2 RMW units).
// No VGPR writeback -> STOREcnt only; DEV scope makes cross-WGP reduction valid.
__device__ inline void atomic_add_f32_dev(float* p, float v) {
  asm volatile("global_atomic_add_f32 %0, %1, off scope:SCOPE_DEV"
               :: "v"(p), "v"(v)
               : "memory");
}

// segment-sum: 32 threads per edge, 4 floats each, f32 atomics into agg
__global__ __launch_bounds__(256) void scatter_add(const float* __restrict__ ef,
                                                   const int* __restrict__ rcv,
                                                   float* __restrict__ agg, int nE) {
  int t = blockIdx.x * 256 + threadIdx.x;
  int e = t >> 5;
  if (e >= nE) return;
  int c = (t & 31) * 4;
  int r = rcv[e];
  const v4f v = *(const v4f*)(ef + (size_t)e * DD + c);
  float* dst = agg + (size_t)r * DD + c;
  atomic_add_f32_dev(dst + 0, v[0]);
  atomic_add_f32_dev(dst + 1, v[1]);
  atomic_add_f32_dev(dst + 2, v[2]);
  atomic_add_f32_dev(dst + 3, v[3]);
}

// ---------------- WMMA MLP kernel ----------------

// A fragment (16x32 f16) from a global f32 row segment of 32 values.
// Lane layout: halves 0..7 = cols hs*8..hs*8+7 ; halves 8..15 = cols 16+hs*8 ..
__device__ inline v16h load_a_frag_g(const float* p, int hs) {
  const v4f* q0 = (const v4f*)(p + hs * 8);
  const v4f* q1 = (const v4f*)(p + 16 + hs * 8);
  v4f x0 = q0[0], x1 = q0[1], y0 = q1[0], y1 = q1[1];
  v16h a;
#pragma unroll
  for (int i = 0; i < 4; i++) {
    a[i]      = (_Float16)x0[i];
    a[4 + i]  = (_Float16)x1[i];
    a[8 + i]  = (_Float16)y0[i];
    a[12 + i] = (_Float16)y1[i];
  }
  return a;
}

__device__ inline v16h load_b_frag_lds(const _Float16* base) {
  const v8h* p = (const v8h*)base;
  v8h b0 = p[0], b1 = p[1];
  v16h b;
#pragma unroll
  for (int i = 0; i < 8; i++) { b[i] = b0[i]; b[8 + i] = b1[i]; }
  return b;
}

struct NodeGather {
  const float* nf; const float* agg;
  __device__ const float* operator()(int r, int kc) const {
    return (kc < 4) ? (nf + (size_t)r * DD + kc * 32)
                    : (agg + (size_t)r * DD + (kc - 4) * 32);
  }
};

struct EdgeGather {
  const float* nn; const float* ef; const int* snd; const int* rcv;
  __device__ const float* operator()(int e, int kc) const {
    if (kc < 4) return nn + (size_t)snd[e] * DD + kc * 32;
    if (kc < 8) return nn + (size_t)rcv[e] * DD + (kc - 4) * 32;
    return ef + (size_t)e * DD + (kc - 8) * 32;
  }
};

// Block = 256 threads = 8 waves; each wave computes 16 rows x 128 cols.
// Layer1: [16 x 32K] x [32K x 128] via v_wmma_f32_16x16x32_f16, KC1 chunks.
// H1 -> LDS (f16) -> re-fragment as A for layer2 (K=128, 4 chunks).
// LayerNorm via half-wave shfl_xor reductions, f32 store.
template <int KC1, class AG>
__global__ __launch_bounds__(256) void mlp_kernel(
    AG ag,
    const _Float16* __restrict__ w1p, const _Float16* __restrict__ w2p,
    const float* __restrict__ bias1, const float* __restrict__ bias2,
    const float* __restrict__ gamma, const float* __restrict__ beta,
    float* __restrict__ out, int nrows) {
  extern __shared__ char smem[];
  _Float16* sW1 = (_Float16*)smem;          // KC1*4096 halves
  _Float16* sW2 = sW1 + KC1 * 4096;         // 16384 halves
  _Float16* sH  = sW2 + 16384;              // 8 waves * 2048 halves

  const int tid = threadIdx.x;
  {  // cooperative stage of packed weights into LDS (b128 traffic)
    const uint4* s1 = (const uint4*)w1p; uint4* d1 = (uint4*)sW1;
    for (int i = tid; i < KC1 * 512; i += 256) d1[i] = s1[i];
    const uint4* s2 = (const uint4*)w2p; uint4* d2 = (uint4*)sW2;
    for (int i = tid; i < 2048; i += 256) d2[i] = s2[i];
  }
  __syncthreads();

  const int wave = tid >> 5, lane = tid & 31;
  const int lrow = lane & 15, hs = lane >> 4;
  const int row  = blockIdx.x * 128 + wave * 16 + lrow;
  const int rowc = row < nrows ? row : nrows - 1;  // clamp loads; keep EXEC full

  v8f acc[8];
#pragma unroll
  for (int t = 0; t < 8; t++)
#pragma unroll
    for (int i = 0; i < 8; i++) acc[t][i] = 0.f;

#pragma unroll 1
  for (int kc = 0; kc < KC1; ++kc) {
    const float* p = ag(rowc, kc);
    v16h a = load_a_frag_g(p, hs);
#pragma unroll
    for (int nt = 0; nt < 8; ++nt) {
      v16h b = load_b_frag_lds(sW1 + (((kc * 8 + nt) * 32 + lane) << 4));
      acc[nt] = __builtin_amdgcn_wmma_f32_16x16x32_f16(
          false, a, false, b, (short)0, acc[nt], false, false);
    }
  }

  // bias1 + convert to f16, stage H1 strip (16x128) in wave-private LDS
  _Float16* hb = sH + wave * 2048;
#pragma unroll
  for (int nt = 0; nt < 8; ++nt) {
    float bv = bias1[nt * 16 + lrow];
#pragma unroll
    for (int j = 0; j < 8; ++j)
      hb[(j + 8 * hs) * DD + nt * 16 + lrow] = (_Float16)(acc[nt][j] + bv);
  }
  __syncthreads();

  v8f acc2[8];
#pragma unroll
  for (int t = 0; t < 8; t++)
#pragma unroll
    for (int i = 0; i < 8; i++) acc2[t][i] = 0.f;

#pragma unroll 1
  for (int kc = 0; kc < 4; ++kc) {
    const v8h* a0 = (const v8h*)(hb + lrow * DD + kc * 32 + hs * 8);
    const v8h* a1 = (const v8h*)(hb + lrow * DD + kc * 32 + 16 + hs * 8);
    v8h x = *a0, y = *a1;
    v16h a;
#pragma unroll
    for (int i = 0; i < 8; i++) { a[i] = x[i]; a[8 + i] = y[i]; }
#pragma unroll
    for (int nt = 0; nt < 8; ++nt) {
      v16h b = load_b_frag_lds(sW2 + (((kc * 8 + nt) * 32 + lane) << 4));
      acc2[nt] = __builtin_amdgcn_wmma_f32_16x16x32_f16(
          false, a, false, b, (short)0, acc2[nt], false, false);
    }
  }

  float b2v[8], gv[8], bev[8];
#pragma unroll
  for (int nt = 0; nt < 8; ++nt) {
    b2v[nt] = bias2[nt * 16 + lrow];
    gv[nt]  = gamma[nt * 16 + lrow];
    bev[nt] = beta[nt * 16 + lrow];
  }

  // LayerNorm: row j+8*hs lives in this 16-lane half across 8 ntiles
#pragma unroll 1
  for (int j = 0; j < 8; ++j) {
    float v[8], s = 0.f, sq = 0.f;
#pragma unroll
    for (int nt = 0; nt < 8; ++nt) {
      float x = acc2[nt][j] + b2v[nt];
      v[nt] = x; s += x; sq += x * x;
    }
#pragma unroll
    for (int m = 1; m < 16; m <<= 1) {
      s  += __shfl_xor(s, m, 32);
      sq += __shfl_xor(sq, m, 32);
    }
    float mu  = s * (1.f / 128.f);
    float var = sq * (1.f / 128.f) - mu * mu;
    float rs  = rsqrtf(var + 1e-5f);
    int orow = blockIdx.x * 128 + wave * 16 + j + 8 * hs;
    if (orow < nrows) {
      float* op = out + (size_t)orow * DD;
#pragma unroll
      for (int nt = 0; nt < 8; ++nt)
        op[nt * 16 + lrow] = (v[nt] - mu) * rs * gv[nt] + bev[nt];
    }
  }
}

// ---------------- launch ----------------

extern "C" void kernel_launch(void* const* d_in, const int* in_sizes, int n_in,
                              void* d_out, int out_size, void* d_ws, size_t ws_size,
                              hipStream_t stream) {
  const float* nf  = (const float*)d_in[0];
  const float* ef  = (const float*)d_in[1];
  const int*   snd = (const int*)d_in[2];
  const int*   rcv = (const int*)d_in[3];
  const float* nW1 = (const float*)d_in[4];
  const float* nb1 = (const float*)d_in[5];
  const float* nW2 = (const float*)d_in[6];
  const float* nb2 = (const float*)d_in[7];
  const float* ng  = (const float*)d_in[8];
  const float* nbb = (const float*)d_in[9];
  const float* eW1 = (const float*)d_in[10];
  const float* eb1 = (const float*)d_in[11];
  const float* eW2 = (const float*)d_in[12];
  const float* eb2 = (const float*)d_in[13];
  const float* eg  = (const float*)d_in[14];
  const float* ebt = (const float*)d_in[15];

  const int nN = in_sizes[0] / DD;
  const int nE = in_sizes[1] / DD;

  float* out_nodes = (float*)d_out;
  float* out_edges = out_nodes + (size_t)nN * DD;

  char* ws = (char*)d_ws;
  float* agg = (float*)ws;
  size_t aggBytes = ((size_t)nN * DD * sizeof(float) + 255) & ~(size_t)255;
  _Float16* nW1p = (_Float16*)(ws + aggBytes);
  _Float16* nW2p = nW1p + 8 * 4096;
  _Float16* eW1p = nW2p + 4 * 4096;
  _Float16* eW2p = eW1p + 12 * 4096;

  { int n4 = nN * DD / 4;
    zero_f4<<<(n4 + 255) / 256, 256, 0, stream>>>((float4*)agg, n4); }

  pack_w<<<(8 * 4096 + 255) / 256, 256, 0, stream>>>(nW1, nW1p, 8 * 4096);
  pack_w<<<(4 * 4096 + 255) / 256, 256, 0, stream>>>(nW2, nW2p, 4 * 4096);
  pack_w<<<(12 * 4096 + 255) / 256, 256, 0, stream>>>(eW1, eW1p, 12 * 4096);
  pack_w<<<(4 * 4096 + 255) / 256, 256, 0, stream>>>(eW2, eW2p, 4 * 4096);

  { int nt = nE * 32;
    scatter_add<<<(nt + 255) / 256, 256, 0, stream>>>(ef, rcv, agg, nE); }

  { int grid = (nN + 127) / 128;
    size_t sh = (size_t)(8 * 4096 + 4 * 4096 + 8 * 2048) * sizeof(_Float16);
    mlp_kernel<8, NodeGather><<<grid, 256, sh, stream>>>(
        NodeGather{nf, agg}, nW1p, nW2p, nb1, nb2, ng, nbb, out_nodes, nN); }

  { int grid = (nE + 127) / 128;
    size_t sh = (size_t)(12 * 4096 + 4 * 4096 + 8 * 2048) * sizeof(_Float16);
    mlp_kernel<12, EdgeGather><<<grid, 256, sh, stream>>>(
        EdgeGather{out_nodes, ef, snd, rcv}, eW1p, eW2p, eb1, eb2, eg, ebt,
        out_edges, nE); }
}
